// CTMCVectorField_18614388260894
// MI455X (gfx1250) — matrix operand order repaired and checked by reading, not compiled, not run on hardware
//
#include <hip/hip_runtime.h>
#include <cstdint>
#include <cstddef>

typedef unsigned short bf16_t;
typedef __attribute__((ext_vector_type(16))) __bf16 v16bf;
typedef __attribute__((ext_vector_type(8)))  float  f32x8;
typedef __attribute__((ext_vector_type(4)))  unsigned u32x4;
typedef __attribute__((ext_vector_type(8)))  int      i32x8;
typedef __attribute__((ext_vector_type(4)))  int      i32x4;

#define WPB   2     // waves per block
#define LANES 32

#if __has_builtin(__builtin_amdgcn_tensor_load_to_lds) && __has_builtin(__builtin_amdgcn_s_wait_tensorcnt)
#define USE_TDM 1
#else
#define USE_TDM 0
#endif

// ---------------- low-level helpers ----------------

__device__ __forceinline__ bf16_t f2bf(float f){
  union { float f; unsigned u; } a; a.f = f;
  unsigned r = a.u + 0x7fffu + ((a.u >> 16) & 1u);   // RNE
  return (bf16_t)(r >> 16);
}

// wave-synchronous LDS ordering: wait for all outstanding DS ops of this wave
__device__ __forceinline__ void lds_fence(){
  __builtin_amdgcn_wave_barrier();
  asm volatile("s_wait_dscnt 0" ::: "memory");
  __builtin_amdgcn_wave_barrier();
}

typedef __attribute__((address_space(3))) const char lds_cchar;
__device__ __forceinline__ unsigned lds_off(const void* p){
  return (unsigned)(size_t)(lds_cchar*)p;    // AS(3) ptrtoint == LDS byte offset
}

__device__ __forceinline__ float fast_sigmoid(float x){
  return __builtin_amdgcn_rcpf(1.f + __expf(-x));    // v_rcp_f32, no div chain
}

union FragU { v16bf v; uint4 q[2]; };

// A-fragment (16x32 bf16) per CDNA5 ISA layout:
// lanes 0-15 (M=lane):  elems 0-7 -> K=kbase..+7,   elems 8-15 -> K=kbase+16..+23
// lanes 16-31 (M=lane-16): elems 0-7 -> K=kbase+8..+15, elems 8-15 -> K=kbase+24..+31
__device__ __forceinline__ v16bf load_a_frag(const bf16_t* abuf, int astride, int kbase, int lane){
  int m  = lane & 15;
  int k0 = kbase + ((lane >> 4) << 3);
  const bf16_t* p = abuf + m * astride + k0;
  FragU f;
  f.q[0] = *reinterpret_cast<const uint4*>(p);
  f.q[1] = *reinterpret_cast<const uint4*>(p + 16);
  return f.v;
}

// B fragments are pre-packed so each lane reads 32 contiguous bytes
__device__ __forceinline__ v16bf load_b_frag(const bf16_t* bpk, int tile, int lane){
  const uint4* p = reinterpret_cast<const uint4*>(bpk) + (size_t)(tile * 32 + lane) * 2;
  FragU f; f.q[0] = p[0]; f.q[1] = p[1];
  return f.v;
}

__device__ __forceinline__ float act_apply(float x, int act){
  if(act == 1) return x * fast_sigmoid(x);  // silu
  if(act == 2) return fast_sigmoid(x);      // sigmoid
  return x;
}

// 16-row tile GEMM: out[16][<=NT*16] = A(16 x KT*32 bf16, LDS) @ Bpacked + bias
// A fragments hoisted into registers and reused across all N tiles.
template<int KT, int NT>
__device__ __forceinline__ void gemm16(const bf16_t* abuf, int astride,
                                       const bf16_t* bpk, const float* bias,
                                       float* outb, int ostride,
                                       int lane, int act, int nvalid){
  v16bf a[KT];
  #pragma unroll
  for(int kt = 0; kt < KT; ++kt) a[kt] = load_a_frag(abuf, astride, kt * 32, lane);
  #pragma unroll
  for(int nt = 0; nt < NT; ++nt){
    f32x8 acc = {};
    #pragma unroll
    for(int kt = 0; kt < KT; ++kt){
      v16bf b = load_b_frag(bpk, nt * KT + kt, lane);
      acc = __builtin_amdgcn_wmma_f32_16x16x32_bf16(false, a[kt], false, b, (short)0, acc, false, false);
    }
    int n = nt * 16 + (lane & 15);
    if(n < nvalid){
      float bv = bias[n];
      #pragma unroll
      for(int r = 0; r < 8; ++r){
        int m = r + ((lane >> 4) << 3);   // C/D layout: VGPR r -> M=r (+8 for hi lanes)
        outb[m * ostride + n] = act_apply(acc[r] + bv, act);
      }
    }
  }
}

__device__ __forceinline__ void f32_to_abuf(const float* src, int sstride,
                                            bf16_t* abuf, int astride, int cols, int lane){
  for(int idx = lane; idx < 16 * cols; idx += LANES){
    int m = idx / cols, c = idx - m * cols;
    abuf[m * astride + c] = f2bf(src[m * sstride + c]);
  }
}

__device__ __forceinline__ void zero_abuf(bf16_t* abuf, int astride, int c0, int c1, int lane){
  int w = c1 - c0;
  if(w <= 0) return;
  for(int idx = lane; idx < 16 * w; idx += LANES){
    int m = idx / w, c = c0 + (idx - m * w);
    abuf[m * astride + c] = 0;
  }
}

__device__ __forceinline__ void layernorm16(float* buf, const float* g, const float* b,
                                            float* tmp, int lane){
  if(lane < 16){
    const float* row = buf + lane * 64;
    float m = 0.f;
    for(int c = 0; c < 64; ++c) m += row[c];
    m *= (1.f / 64.f);
    float v = 0.f;
    for(int c = 0; c < 64; ++c){ float t = row[c] - m; v += t * t; }
    v *= (1.f / 64.f);
    tmp[lane * 2 + 0] = m;
    tmp[lane * 2 + 1] = __builtin_amdgcn_rsqf(v + 1e-5f);
  }
  lds_fence();
  for(int idx = lane; idx < 16 * 64; idx += LANES){
    int m = idx >> 6, c = idx & 63;
    buf[idx] = (buf[idx] - tmp[m * 2]) * tmp[m * 2 + 1] * g[c] + b[c];
  }
  lds_fence();
}

// ---------------- GVP ----------------

struct GvpDev {
  const float*  Wh;
  const bf16_t* Ws;  const float* Wsb;
  const float*  Wv;
  const bf16_t* Wg;  const float* Wgb;
};

// vbuf stride is fixed at 17 vector slots x 3 comps per row
template<int SIN, int VIN, int VOUT, int HD>
__device__ __forceinline__ void gvp_run(const GvpDev g, bf16_t* abuf, int astride,
                                        float* vbuf, float* vhbuf,
                                        float* soutb, float* gateb, int lane){
  constexpr int KT = (SIN + HD + 31) / 32;
  // Vh[n][h][c] = sum_v V[n][v][c] * Wh[v][h]
  for(int idx = lane; idx < 16 * HD * 3; idx += LANES){
    int n = idx / (HD * 3), r = idx - n * (HD * 3), h = r / 3, c = r - h * 3;
    float acc = 0.f;
    #pragma unroll
    for(int v = 0; v < VIN; ++v) acc += vbuf[(n * 17 + v) * 3 + c] * g.Wh[v * HD + h];
    vhbuf[(n * HD + h) * 3 + c] = acc;
  }
  lds_fence();
  // vn -> abuf[:, SIN : SIN+HD] ; zero-pad rest of K window
  for(int idx = lane; idx < 16 * HD; idx += LANES){
    int n = idx / HD, h = idx - n * HD;
    float s2 = 1e-8f;
    #pragma unroll
    for(int c = 0; c < 3; ++c){ float t = vhbuf[(n * HD + h) * 3 + c]; s2 += t * t; }
    abuf[n * astride + SIN + h] = f2bf(sqrtf(s2));
  }
  zero_abuf(abuf, astride, SIN + HD, KT * 32, lane);
  lds_fence();
  // so = silu([s, vn] @ Ws + b)
  gemm16<KT, 4>(abuf, astride, g.Ws, g.Wsb, soutb, 64, lane, 1, 64);
  lds_fence();
  // Vo[n][u][c] = sum_h Vh[n][h][c] * Wv[h][u]   (overwrites vbuf)
  for(int idx = lane; idx < 16 * VOUT * 3; idx += LANES){
    int n = idx / (VOUT * 3), r = idx - n * (VOUT * 3), u = r / 3, c = r - u * 3;
    float acc = 0.f;
    #pragma unroll
    for(int h = 0; h < HD; ++h) acc += vhbuf[(n * HD + h) * 3 + c] * g.Wv[h * VOUT + u];
    vbuf[(n * 17 + u) * 3 + c] = acc;
  }
  // so -> abuf (bf16) and gate = sigmoid(so @ Wg + b)
  f32_to_abuf(soutb, 64, abuf, astride, 64, lane);
  lds_fence();
  gemm16<2, 1>(abuf, astride, g.Wg, g.Wgb, gateb, 16, lane, 2, VOUT);
  lds_fence();
  for(int idx = lane; idx < 16 * VOUT * 3; idx += LANES){
    int n = idx / (VOUT * 3), r = idx - n * (VOUT * 3), u = r / 3;
    vbuf[(n * 17 + u) * 3 + (r - u * 3)] *= gateb[n * 16 + u];
  }
  lds_fence();
}

// ---------------- kernels ----------------

// repack weight w[K][N] (fp32 row-major) into WMMA B fragments (bf16), 1KB per 32x16 tile
__global__ __launch_bounds__(32) void repack_b_kernel(const float* w, bf16_t* dst,
                                                      int K, int N, int nKt){
  int tile = blockIdx.x, lane = threadIdx.x;
  int kt = tile % nKt, nt = tile / nKt;
  int n  = nt * 16 + (lane & 15);
  int kb = kt * 32 + ((lane >> 4) << 3);  // B layout: k = (e<8 ? e : e+8) + 8*(lane>=16)
  __align__(16) bf16_t tmp[16];
  #pragma unroll
  for(int e = 0; e < 16; ++e){
    int k = kb + (e < 8 ? e : e + 8);
    float v = (k < K && n < N) ? w[(size_t)k * N + n] : 0.f;
    tmp[e] = f2bf(v);
  }
  uint4* d = reinterpret_cast<uint4*>(dst) + (size_t)(tile * 32 + lane) * 2;
  d[0] = *reinterpret_cast<uint4*>(&tmp[0]);
  d[1] = *reinterpret_cast<uint4*>(&tmp[8]);
}

struct EmbedNodeArgs {
  const float* a_t; const float* c_t; const float* tvec; const int* nbi;
  const bf16_t* w0; const float* b0; const bf16_t* w1; const float* b1;
  const float* lng; const float* lnb; float* s_out; int N;
};

__global__ __launch_bounds__(WPB*32) void embed_node_kernel(EmbedNodeArgs A){
  __shared__ __align__(16) bf16_t abuf_s[WPB][16*64];
  __shared__ float soutb_s[WPB][16*64];
  __shared__ float tmpb_s[WPB][32];
  int wave = threadIdx.x >> 5, lane = threadIdx.x & 31;
  bf16_t* abuf = abuf_s[wave]; float* soutb = soutb_s[wave]; float* tmpb = tmpb_s[wave];
  long n0 = ((long)blockIdx.x * WPB + wave) * 16;
  if(n0 >= A.N) return;
  for(int idx = lane; idx < 16 * 32; idx += LANES){
    int m = idx >> 5, c = idx & 31;
    long r = n0 + m; if(r >= A.N) r = A.N - 1;
    float v = 0.f;
    if(c < 11)       v = A.a_t[r * 11 + c];
    else if(c == 11) v = A.tvec[A.nbi[r]];
    else if(c < 19)  v = A.c_t[r * 7 + (c - 12)];
    abuf[m * 64 + c] = f2bf(v);
  }
  lds_fence();
  gemm16<1, 4>(abuf, 64, A.w0, A.b0, soutb, 64, lane, 1, 64);
  lds_fence();
  f32_to_abuf(soutb, 64, abuf, 64, 64, lane);
  lds_fence();
  gemm16<2, 4>(abuf, 64, A.w1, A.b1, soutb, 64, lane, 1, 64);
  lds_fence();
  layernorm16(soutb, A.lng, A.lnb, tmpb, lane);
  for(int idx = lane; idx < 16 * 64; idx += LANES){
    long r = n0 + (idx >> 6);
    if(r < A.N) A.s_out[r * 64 + (idx & 63)] = soutb[idx];
  }
}

struct EmbedEdgeArgs {
  const float* e_t;
  const bf16_t* w0; const float* b0; const bf16_t* w1; const float* b1;
  const float* lng; const float* lnb; float* ef_out; int E;
};

__global__ __launch_bounds__(WPB*32) void embed_edge_kernel(EmbedEdgeArgs A){
  __shared__ __align__(16) bf16_t abuf_s[WPB][16*64];
  __shared__ float soutb_s[WPB][16*64];
  __shared__ float tmpb_s[WPB][32];
  int wave = threadIdx.x >> 5, lane = threadIdx.x & 31;
  bf16_t* abuf = abuf_s[wave]; float* soutb = soutb_s[wave]; float* tmpb = tmpb_s[wave];
  long e0 = ((long)blockIdx.x * WPB + wave) * 16;
  if(e0 >= A.E) return;
  for(int idx = lane; idx < 16 * 32; idx += LANES){
    int m = idx >> 5, c = idx & 31;
    long r = e0 + m; if(r >= A.E) r = A.E - 1;
    float v = (c < 6) ? A.e_t[r * 6 + c] : 0.f;
    abuf[m * 64 + c] = f2bf(v);
  }
  lds_fence();
  gemm16<1, 4>(abuf, 64, A.w0, A.b0, soutb, 64, lane, 1, 64);
  lds_fence();
  f32_to_abuf(soutb, 64, abuf, 64, 64, lane);
  lds_fence();
  gemm16<2, 4>(abuf, 64, A.w1, A.b1, soutb, 64, lane, 1, 64);
  lds_fence();
  layernorm16(soutb, A.lng, A.lnb, tmpb, lane);
  for(int idx = lane; idx < 16 * 64; idx += LANES){
    long r = e0 + (idx >> 6);
    if(r < A.E) A.ef_out[r * 64 + (idx & 63)] = soutb[idx];
  }
}

struct EdgeMsgArgs {
  const float* s; const float* V; const float* x; const float* ef;
  const int* esrc; const int* edst;
  float* agg_s; float* agg_V;
  GvpDev g0, g1, g2;
  int E;
};

__global__ __launch_bounds__(WPB*32) void edge_msg_kernel(EdgeMsgArgs A){
  __shared__ __align__(16) bf16_t abuf_s[WPB][16*192];
  __shared__ float vbuf_s[WPB][16*17*3];
  __shared__ float vhbuf_s[WPB][16*17*3];
  __shared__ float soutb_s[WPB][16*64];
  __shared__ float gateb_s[WPB][16*16];
  __shared__ int   idxb_s[WPB][32];
#if USE_TDM
  __shared__ __align__(16) float efstg_s[WPB][16*64];
#endif
  int wave = threadIdx.x >> 5, lane = threadIdx.x & 31;
  bf16_t* abuf = abuf_s[wave]; float* vbuf = vbuf_s[wave]; float* vhbuf = vhbuf_s[wave];
  float* soutb = soutb_s[wave]; float* gateb = gateb_s[wave]; int* idxb = idxb_s[wave];
  long e0 = ((long)blockIdx.x * WPB + wave) * 16;
  if(e0 >= A.E) return;

#if USE_TDM
  // Async TDM: copy this wave's contiguous 16x64 fp32 ef tile (4KB) into LDS staging.
  // D# group0: count=1 | lds_addr | 57-bit global addr | type=2 ("image")
  // D# group1: data_size=4B, 1-D tensor: tensor_dim0 = tile_dim0 = 1024 elems, stride 1024
  float* efstg = efstg_s[wave];
  {
    unsigned long long ga = (unsigned long long)(const void*)(A.ef + (size_t)e0 * 64);
    u32x4 g0;
    g0[0] = 1u;                                            // count=1 (valid descriptor)
    g0[1] = lds_off(efstg);                                // lds_addr (bytes)
    g0[2] = (unsigned)(ga & 0xffffffffu);                  // global_addr[31:0]
    g0[3] = (unsigned)((ga >> 32) & 0x01ffffffu) | (2u << 30); // addr[56:32] | type=2
    i32x8 g1;
    g1[0] = (int)(2u << 16);        // workgroup_mask=0, data_size=2 (4 bytes)
    g1[1] = (int)(1024u << 16);     // tensor_dim0[15:0] = 1024
    g1[2] = (int)(1u << 16);        // tensor_dim0 hi = 0, tensor_dim1[15:0] = 1
    g1[3] = (int)(1024u << 16);     // tensor_dim1 hi = 0, tile_dim0 = 1024
    g1[4] = 0;                      // tile_dim1 = 0 (unused), tile_dim2 = 0
    g1[5] = 1024;                   // tensor_dim0_stride lo32
    g1[6] = 0;
    g1[7] = 0;
    i32x4 z4 = {0, 0, 0, 0};
    i32x8 z8 = {0, 0, 0, 0, 0, 0, 0, 0};
    __builtin_amdgcn_tensor_load_to_lds(g0, g1, z4, z4, z8, 0);
  }
#endif

  int row = lane & 15;
  long e = e0 + row; if(e >= A.E) e = A.E - 1;
  idxb[lane] = (lane < 16) ? A.esrc[e] : A.edst[e];
  lds_fence();
  if(lane < 16){
    int si = idxb[row], di = idxb[16 + row];
    float dx0 = A.x[(size_t)di*3+0] - A.x[(size_t)si*3+0];
    float dx1 = A.x[(size_t)di*3+1] - A.x[(size_t)si*3+1];
    float dx2 = A.x[(size_t)di*3+2] - A.x[(size_t)si*3+2];
    float d = sqrtf(dx0*dx0 + dx1*dx1 + dx2*dx2 + 1e-8f);
    float inv = __builtin_amdgcn_rcpf(d);
    vbuf[(row*17 + 16)*3 + 0] = dx0 * inv;
    vbuf[(row*17 + 16)*3 + 1] = dx1 * inv;
    vbuf[(row*17 + 16)*3 + 2] = dx2 * inv;
    #pragma unroll
    for(int j = 0; j < 16; ++j){
      float mu = (20.f / 15.f) * (float)j;   // linspace(0, 20, 16)
      float z  = (d - mu) * (16.f / 20.f);   // sigma = 20/16
      abuf[row*192 + 128 + j] = f2bf(__expf(-z * z));
    }
  }
  // gather s[src] (L2-resident) while the TDM streams ef
  for(int idx = lane; idx < 16 * 64; idx += LANES){
    int m = idx >> 6, c = idx & 63;
    abuf[m*192 + c] = f2bf(A.s[(size_t)idxb[m]*64 + c]);
  }
  for(int idx = lane; idx < 16 * 48; idx += LANES){
    int m = idx / 48, r = idx - m * 48;
    vbuf[m*17*3 + r] = A.V[(size_t)idxb[m]*48 + r];
  }
#if USE_TDM
  __builtin_amdgcn_s_wait_tensorcnt(0);
  __builtin_amdgcn_wave_barrier();
  for(int idx = lane; idx < 16 * 64; idx += LANES){
    int m = idx >> 6, c = idx & 63;
    abuf[m*192 + 64 + c] = f2bf(efstg[m*64 + c]);
  }
#else
  for(int idx = lane; idx < 16 * 64; idx += LANES){
    int m = idx >> 6, c = idx & 63;
    long ee = e0 + m; if(ee >= A.E) ee = A.E - 1;
    abuf[m*192 + 64 + c] = f2bf(A.ef[(size_t)ee*64 + c]);
  }
#endif
  lds_fence();
  gvp_run<144, 17, 16, 17>(A.g0, abuf, 192, vbuf, vhbuf, soutb, gateb, lane);
  gvp_run< 64, 16, 16, 16>(A.g1, abuf, 192, vbuf, vhbuf, soutb, gateb, lane);
  gvp_run< 64, 16, 16, 16>(A.g2, abuf, 192, vbuf, vhbuf, soutb, gateb, lane);
  for(int idx = lane; idx < 16 * 64; idx += LANES){
    int m = idx >> 6, c = idx & 63;
    if(e0 + m < A.E)
      atomicAdd(&A.agg_s[(size_t)idxb[16+m]*64 + c], soutb[m*64 + c]);
  }
  for(int idx = lane; idx < 16 * 48; idx += LANES){
    int m = idx / 48, r = idx - m * 48; int u = r / 3, c = r - u * 3;
    if(e0 + m < A.E)
      atomicAdd(&A.agg_V[(size_t)idxb[16+m]*48 + r], vbuf[(m*17 + u)*3 + c]);
  }
}

struct NodeUpdArgs {
  float* s; float* V; const float* agg_s; const float* agg_V;
  GvpDev u0, u1, u2;
  const float* ln1g; const float* ln1b; const float* ln2g; const float* ln2b;
  int N;
};

__global__ __launch_bounds__(WPB*32) void node_upd_kernel(NodeUpdArgs A){
  __shared__ __align__(16) bf16_t abuf_s[WPB][16*96];
  __shared__ float vbuf_s[WPB][16*17*3];
  __shared__ float vhbuf_s[WPB][16*17*3];
  __shared__ float soutb_s[WPB][16*64];
  __shared__ float sresb_s[WPB][16*64];
  __shared__ float vresb_s[WPB][16*48];
  __shared__ float gateb_s[WPB][16*16];
  int wave = threadIdx.x >> 5, lane = threadIdx.x & 31;
  bf16_t* abuf = abuf_s[wave]; float* vbuf = vbuf_s[wave]; float* vhbuf = vhbuf_s[wave];
  float* soutb = soutb_s[wave]; float* sresb = sresb_s[wave];
  float* vresb = vresb_s[wave]; float* gateb = gateb_s[wave];
  long n0 = ((long)blockIdx.x * WPB + wave) * 16;
  if(n0 >= A.N) return;
  for(int idx = lane; idx < 16 * 64; idx += LANES){
    int m = idx >> 6, c = idx & 63;
    long r = n0 + m; if(r >= A.N) r = A.N - 1;
    soutb[idx] = A.s[(size_t)r*64 + c] + 0.01f * A.agg_s[(size_t)r*64 + c];
  }
  lds_fence();
  layernorm16(soutb, A.ln1g, A.ln1b, gateb, lane);
  for(int idx = lane; idx < 16 * 64; idx += LANES) sresb[idx] = soutb[idx];
  f32_to_abuf(soutb, 64, abuf, 96, 64, lane);
  for(int idx = lane; idx < 16 * 48; idx += LANES){
    int m = idx / 48, r = idx - m * 48; int u = r / 3, c = r - u * 3;
    long ri = n0 + m; if(ri >= A.N) ri = A.N - 1;
    float v = A.V[(size_t)ri*48 + r] + 0.01f * A.agg_V[(size_t)ri*48 + r];
    vresb[m*48 + r] = v;
    vbuf[(m*17 + u)*3 + c] = v;
  }
  lds_fence();
  gvp_run<64, 16, 16, 16>(A.u0, abuf, 96, vbuf, vhbuf, soutb, gateb, lane);
  gvp_run<64, 16, 16, 16>(A.u1, abuf, 96, vbuf, vhbuf, soutb, gateb, lane);
  gvp_run<64, 16, 16, 16>(A.u2, abuf, 96, vbuf, vhbuf, soutb, gateb, lane);
  for(int idx = lane; idx < 16 * 64; idx += LANES) soutb[idx] += sresb[idx];
  lds_fence();
  layernorm16(soutb, A.ln2g, A.ln2b, gateb, lane);
  for(int idx = lane; idx < 16 * 64; idx += LANES){
    int m = idx >> 6, c = idx & 63;
    long r = n0 + m;
    if(r < A.N) A.s[(size_t)r*64 + c] = soutb[idx];
  }
  for(int idx = lane; idx < 16 * 48; idx += LANES){
    int m = idx / 48, r = idx - m * 48; int u = r / 3, c = r - u * 3;
    long ri = n0 + m;
    if(ri < A.N) A.V[(size_t)ri*48 + r] = vresb[m*48 + r] + vbuf[(m*17 + u)*3 + c];
  }
}

struct PosArgs { const float* s; const float* V; float* x; GvpDev p0, p1, p2; int N; };

__global__ __launch_bounds__(WPB*32) void pos_upd_kernel(PosArgs A){
  __shared__ __align__(16) bf16_t abuf_s[WPB][16*96];
  __shared__ float vbuf_s[WPB][16*17*3];
  __shared__ float vhbuf_s[WPB][16*17*3];
  __shared__ float soutb_s[WPB][16*64];
  __shared__ float gateb_s[WPB][16*16];
  int wave = threadIdx.x >> 5, lane = threadIdx.x & 31;
  bf16_t* abuf = abuf_s[wave]; float* vbuf = vbuf_s[wave]; float* vhbuf = vhbuf_s[wave];
  float* soutb = soutb_s[wave]; float* gateb = gateb_s[wave];
  long n0 = ((long)blockIdx.x * WPB + wave) * 16;
  if(n0 >= A.N) return;
  for(int idx = lane; idx < 16 * 64; idx += LANES){
    int m = idx >> 6, c = idx & 63;
    long r = n0 + m; if(r >= A.N) r = A.N - 1;
    abuf[m*96 + c] = f2bf(A.s[(size_t)r*64 + c]);
  }
  for(int idx = lane; idx < 16 * 48; idx += LANES){
    int m = idx / 48, r = idx - m * 48; int u = r / 3, c = r - u * 3;
    long ri = n0 + m; if(ri >= A.N) ri = A.N - 1;
    vbuf[(m*17 + u)*3 + c] = A.V[(size_t)ri*48 + r];
  }
  lds_fence();
  gvp_run<64, 16, 16, 16>(A.p0, abuf, 96, vbuf, vhbuf, soutb, gateb, lane);
  gvp_run<64, 16, 16, 16>(A.p1, abuf, 96, vbuf, vhbuf, soutb, gateb, lane);
  gvp_run<64, 16,  1, 16>(A.p2, abuf, 96, vbuf, vhbuf, soutb, gateb, lane);
  if(lane < 16){
    long r = n0 + lane;
    if(r < A.N){
      #pragma unroll
      for(int c = 0; c < 3; ++c)
        A.x[(size_t)r*3 + c] += vbuf[(lane*17 + 0)*3 + c];
    }
  }
}

struct EdgeUpdArgs {
  const float* s; float* ef; const int* esrc; const int* edst;
  const bf16_t* w1; const float* b1; const bf16_t* w2; const float* b2;
  const float* lng; const float* lnb; int E;
};

__global__ __launch_bounds__(WPB*32) void edge_upd_kernel(EdgeUpdArgs A){
  __shared__ __align__(16) bf16_t abuf_s[WPB][16*192];
  __shared__ float soutb_s[WPB][16*64];
  __shared__ float resb_s[WPB][16*64];
  __shared__ float tmpb_s[WPB][32];
  __shared__ int   idxb_s[WPB][32];
  int wave = threadIdx.x >> 5, lane = threadIdx.x & 31;
  bf16_t* abuf = abuf_s[wave]; float* soutb = soutb_s[wave];
  float* resb = resb_s[wave]; float* tmpb = tmpb_s[wave]; int* idxb = idxb_s[wave];
  long e0 = ((long)blockIdx.x * WPB + wave) * 16;
  if(e0 >= A.E) return;
  int row = lane & 15;
  long e = e0 + row; if(e >= A.E) e = A.E - 1;
  idxb[lane] = (lane < 16) ? A.esrc[e] : A.edst[e];
  lds_fence();
  for(int idx = lane; idx < 16 * 64; idx += LANES){
    int m = idx >> 6, c = idx & 63;
    long ee = e0 + m; if(ee >= A.E) ee = A.E - 1;
    float efv = A.ef[(size_t)ee*64 + c];
    abuf[m*192 + c]       = f2bf(A.s[(size_t)idxb[m]*64 + c]);
    abuf[m*192 + 64 + c]  = f2bf(A.s[(size_t)idxb[16+m]*64 + c]);
    abuf[m*192 + 128 + c] = f2bf(efv);
    resb[idx] = efv;
  }
  lds_fence();
  gemm16<6, 4>(abuf, 192, A.w1, A.b1, soutb, 64, lane, 1, 64);
  lds_fence();
  f32_to_abuf(soutb, 64, abuf, 192, 64, lane);
  lds_fence();
  gemm16<2, 4>(abuf, 192, A.w2, A.b2, soutb, 64, lane, 1, 64);
  lds_fence();
  for(int idx = lane; idx < 16 * 64; idx += LANES) soutb[idx] += resb[idx];
  lds_fence();
  layernorm16(soutb, A.lng, A.lnb, tmpb, lane);
  for(int idx = lane; idx < 16 * 64; idx += LANES){
    long ee = e0 + (idx >> 6);
    if(ee < A.E) A.ef[(size_t)ee*64 + (idx & 63)] = soutb[idx];
  }
}

struct HeadArgs {
  const float* in; float* out;
  const bf16_t* w0; const float* b0; const bf16_t* w1; const float* b1;
  int n; int ocols;
};

__global__ __launch_bounds__(WPB*32) void head_kernel(HeadArgs A){
  __shared__ __align__(16) bf16_t abuf_s[WPB][16*64];
  __shared__ float soutb_s[WPB][16*64];
  int wave = threadIdx.x >> 5, lane = threadIdx.x & 31;
  bf16_t* abuf = abuf_s[wave]; float* soutb = soutb_s[wave];
  long n0 = ((long)blockIdx.x * WPB + wave) * 16;
  if(n0 >= A.n) return;
  for(int idx = lane; idx < 16 * 64; idx += LANES){
    int m = idx >> 6, c = idx & 63;
    long r = n0 + m; if(r >= A.n) r = A.n - 1;
    abuf[m*64 + c] = f2bf(A.in[(size_t)r*64 + c]);
  }
  lds_fence();
  gemm16<2, 4>(abuf, 64, A.w0, A.b0, soutb, 64, lane, 1, 64);
  lds_fence();
  f32_to_abuf(soutb, 64, abuf, 64, 64, lane);
  lds_fence();
  gemm16<2, 1>(abuf, 64, A.w1, A.b1, soutb, 64, lane, 0, A.ocols);
  lds_fence();
  for(int idx = lane; idx < 16 * A.ocols; idx += LANES){
    int m = idx / A.ocols, c = idx - m * A.ocols;
    long r = n0 + m;
    if(r < A.n) A.out[(size_t)r * A.ocols + c] = soutb[m*64 + c];
  }
}

// ---------------- host side ----------------

namespace {
struct LinOff { int b, w; };
struct LnOff  { int b, g; };
struct GvpOff { LinOff Wg; int Wh; LinOff Ws; int Wv; };
}

extern "C" void kernel_launch(void* const* d_in, const int* in_sizes, int n_in,
                              void* d_out, int out_size, void* d_ws, size_t ws_size,
                              hipStream_t stream){
  (void)n_in; (void)out_size; (void)ws_size;
  const float* a_t  = (const float*)d_in[0];
  const float* c_t  = (const float*)d_in[1];
  const float* x_t  = (const float*)d_in[2];
  const float* e_t  = (const float*)d_in[3];
  const float* tvec = (const float*)d_in[4];
  const float* prm  = (const float*)d_in[5];
  const int*   nbi  = (const int*)d_in[6];
  const int*   esrc = (const int*)d_in[7];
  const int*   edst = (const int*)d_in[8];
  const int N = in_sizes[0] / 11;   // A+1 = 11
  const int E = in_sizes[3] / 6;    // BT+1 = 6

  // --- parameter offsets (pytree flatten: dict keys alphabetical, lists in order) ---
  int cur = 0;
  auto take  = [&](int n){ int o = cur; cur += n; return o; };
  auto mklin = [&](int i, int o){ LinOff l; l.b = take(o); l.w = take(i * o); return l; };
  auto mkln  = [&](int d){ LnOff l; l.b = take(d); l.g = take(d); return l; };
  auto mkgvp = [&](int si, int vi, int so, int vo){
    int h = vi > vo ? vi : vo;
    GvpOff g; g.Wg = mklin(so, vo); g.Wh = take(vi * h); g.Ws = mklin(si + h, so); g.Wv = take(h * vo);
    return g;
  };
  // top-level keys: convs, edge_emb, edge_head, edge_upd, node_head, pos_upd, scalar_emb
  LnOff ln1[4], ln2[4]; GvpOff msg[4][3], upd[4][3];
  for(int l = 0; l < 4; ++l){       // per conv keys: ln1, ln2, msg, upd
    ln1[l] = mkln(64); ln2[l] = mkln(64);
    msg[l][0] = mkgvp(144, 17, 64, 16);
    msg[l][1] = mkgvp(64, 16, 64, 16);
    msg[l][2] = mkgvp(64, 16, 64, 16);
    for(int j = 0; j < 3; ++j) upd[l][j] = mkgvp(64, 16, 64, 16);
  }
  LinOff ee0 = mklin(6, 64),  ee1 = mklin(64, 64); LnOff eeln = mkln(64);
  LinOff eh0 = mklin(64, 64), eh1 = mklin(64, 5);
  LinOff eu1 = mklin(192, 64), eu2 = mklin(64, 64); LnOff euln = mkln(64);
  LinOff nh0 = mklin(64, 64), nh1 = mklin(64, 16);
  GvpOff pu[3];
  pu[0] = mkgvp(64, 16, 64, 16); pu[1] = mkgvp(64, 16, 64, 16); pu[2] = mkgvp(64, 16, 64, 1);
  LinOff se0 = mklin(19, 64), se1 = mklin(64, 64); LnOff seln = mkln(64);

  // --- workspace carving ---
  char* ws = (char*)d_ws;
  size_t wcur = 0;
  auto carve = [&](size_t bytes)->char*{
    wcur = (wcur + 255) & ~(size_t)255;
    char* p = ws + wcur; wcur += bytes; return p;
  };
  float* sBuf  = (float*)carve((size_t)N * 64 * 4);
  float* VBuf  = (float*)carve((size_t)N * 48 * 4);
  float* efBuf = (float*)carve((size_t)E * 64 * 4);
  float* xBuf  = (float*)carve((size_t)N * 3 * 4);
  float* aggS  = (float*)carve((size_t)N * 64 * 4);
  float* aggV  = (float*)carve((size_t)N * 48 * 4);

  auto pack = [&](int off, int K, int Nc)->const bf16_t*{
    int nKt = (K + 31) / 32, nNt = (Nc + 15) / 16;
    bf16_t* dst = (bf16_t*)carve((size_t)nKt * nNt * 512 * sizeof(bf16_t));
    repack_b_kernel<<<nKt * nNt, 32, 0, stream>>>(prm + off, dst, K, Nc, nKt);
    return dst;
  };
  auto mkdev = [&](const GvpOff& g, int si, int vi, int so, int vo)->GvpDev{
    int h = vi > vo ? vi : vo;
    GvpDev d;
    d.Wh  = prm + g.Wh;
    d.Ws  = pack(g.Ws.w, si + h, so); d.Wsb = prm + g.Ws.b;
    d.Wv  = prm + g.Wv;
    d.Wg  = pack(g.Wg.w, so, vo);     d.Wgb = prm + g.Wg.b;
    return d;
  };

  GvpDev msgD[4][3], updD[4][3], puD[3];
  for(int l = 0; l < 4; ++l){
    msgD[l][0] = mkdev(msg[l][0], 144, 17, 64, 16);
    msgD[l][1] = mkdev(msg[l][1], 64, 16, 64, 16);
    msgD[l][2] = mkdev(msg[l][2], 64, 16, 64, 16);
    for(int j = 0; j < 3; ++j) updD[l][j] = mkdev(upd[l][j], 64, 16, 64, 16);
  }
  puD[0] = mkdev(pu[0], 64, 16, 64, 16);
  puD[1] = mkdev(pu[1], 64, 16, 64, 16);
  puD[2] = mkdev(pu[2], 64, 16, 64, 1);
  const bf16_t* se0w = pack(se0.w, 19, 64);
  const bf16_t* se1w = pack(se1.w, 64, 64);
  const bf16_t* ee0w = pack(ee0.w, 6, 64);
  const bf16_t* ee1w = pack(ee1.w, 64, 64);
  const bf16_t* eu1w = pack(eu1.w, 192, 64);
  const bf16_t* eu2w = pack(eu2.w, 64, 64);
  const bf16_t* nh0w = pack(nh0.w, 64, 64);
  const bf16_t* nh1w = pack(nh1.w, 64, 16);
  const bf16_t* eh0w = pack(eh0.w, 64, 64);
  const bf16_t* eh1w = pack(eh1.w, 64, 5);

  (void)hipMemsetAsync(VBuf, 0, (size_t)N * 48 * 4, stream);
  (void)hipMemcpyAsync(xBuf, x_t, (size_t)N * 3 * 4, hipMemcpyDeviceToDevice, stream);

  const int nTN = (N + 15) / 16, nTE = (E + 15) / 16;
  const int gbN = (nTN + WPB - 1) / WPB, gbE = (nTE + WPB - 1) / WPB;

  {
    EmbedNodeArgs a;
    a.a_t = a_t; a.c_t = c_t; a.tvec = tvec; a.nbi = nbi;
    a.w0 = se0w; a.b0 = prm + se0.b; a.w1 = se1w; a.b1 = prm + se1.b;
    a.lng = prm + seln.g; a.lnb = prm + seln.b; a.s_out = sBuf; a.N = N;
    embed_node_kernel<<<gbN, WPB * 32, 0, stream>>>(a);
  }
  {
    EmbedEdgeArgs a;
    a.e_t = e_t;
    a.w0 = ee0w; a.b0 = prm + ee0.b; a.w1 = ee1w; a.b1 = prm + ee1.b;
    a.lng = prm + eeln.g; a.lnb = prm + eeln.b; a.ef_out = efBuf; a.E = E;
    embed_edge_kernel<<<gbE, WPB * 32, 0, stream>>>(a);
  }

  for(int l = 0; l < 4; ++l){
    (void)hipMemsetAsync(aggS, 0, (size_t)N * 64 * 4, stream);
    (void)hipMemsetAsync(aggV, 0, (size_t)N * 48 * 4, stream);
    {
      EdgeMsgArgs a;
      a.s = sBuf; a.V = VBuf; a.x = xBuf; a.ef = efBuf;
      a.esrc = esrc; a.edst = edst; a.agg_s = aggS; a.agg_V = aggV;
      a.g0 = msgD[l][0]; a.g1 = msgD[l][1]; a.g2 = msgD[l][2]; a.E = E;
      edge_msg_kernel<<<gbE, WPB * 32, 0, stream>>>(a);
    }
    {
      NodeUpdArgs a;
      a.s = sBuf; a.V = VBuf; a.agg_s = aggS; a.agg_V = aggV;
      a.u0 = updD[l][0]; a.u1 = updD[l][1]; a.u2 = updD[l][2];
      a.ln1g = prm + ln1[l].g; a.ln1b = prm + ln1[l].b;
      a.ln2g = prm + ln2[l].g; a.ln2b = prm + ln2[l].b; a.N = N;
      node_upd_kernel<<<gbN, WPB * 32, 0, stream>>>(a);
    }
    if((l & 1) == 1){
      PosArgs a;
      a.s = sBuf; a.V = VBuf; a.x = xBuf;
      a.p0 = puD[0]; a.p1 = puD[1]; a.p2 = puD[2]; a.N = N;
      pos_upd_kernel<<<gbN, WPB * 32, 0, stream>>>(a);
      EdgeUpdArgs b;
      b.s = sBuf; b.ef = efBuf; b.esrc = esrc; b.edst = edst;
      b.w1 = eu1w; b.b1 = prm + eu1.b; b.w2 = eu2w; b.b2 = prm + eu2.b;
      b.lng = prm + euln.g; b.lnb = prm + euln.b; b.E = E;
      edge_upd_kernel<<<gbE, WPB * 32, 0, stream>>>(b);
    }
  }

  float* out = (float*)d_out;
  {
    HeadArgs a;
    a.in = sBuf; a.out = out;
    a.w0 = nh0w; a.b0 = prm + nh0.b; a.w1 = nh1w; a.b1 = prm + nh1.b;
    a.n = N; a.ocols = 16;
    head_kernel<<<gbN, WPB * 32, 0, stream>>>(a);
  }
  {
    HeadArgs a;
    a.in = efBuf; a.out = out + (size_t)N * 16;
    a.w0 = eh0w; a.b0 = prm + eh0.b; a.w1 = eh1w; a.b1 = prm + eh1.b;
    a.n = E; a.ocols = 5;
    head_kernel<<<gbE, WPB * 32, 0, stream>>>(a);
  }
  (void)hipMemcpyAsync(out + (size_t)N * 16 + (size_t)E * 5, xBuf,
                       (size_t)N * 3 * 4, hipMemcpyDeviceToDevice, stream);
}
